// PointDecompositionModel_64622077935824
// MI455X (gfx1250) — compile-verified, matrix operand order checked by reading, not run on hardware
//
#include <hip/hip_runtime.h>

typedef __attribute__((ext_vector_type(16))) _Float16 v16h;
typedef __attribute__((ext_vector_type(8)))  _Float16 v8h;
typedef __attribute__((ext_vector_type(8)))  float    v8f;

#define EPS_BN 1e-5f

// ============================================================================
// Weight pad + f32 -> f16 convert.  wp[n, Kp] = (k < K) ? w[n, K] : 0
// ============================================================================
__global__ void padw_kernel(const float* __restrict__ w, _Float16* __restrict__ wp,
                            int N, int K, int Kp) {
  int t = blockIdx.x * 256 + threadIdx.x;
  if (t >= N * Kp) return;
  int n = t / Kp, k = t - n * Kp;
  wp[t] = (_Float16)(k < K ? w[n * K + k] : 0.0f);
}

// ============================================================================
// Farthest point sampling: one workgroup per batch, dist array lives in LDS.
// ============================================================================
__global__ __launch_bounds__(256)
void fps_kernel(const float* __restrict__ xyz, int* __restrict__ fidx,
                int N, int npoint) {
  __shared__ float dist[8192];
  __shared__ float rmax[256];
  __shared__ int   rloc[256];
  __shared__ float cen[3];
  __shared__ int   sfar;
  const int b = blockIdx.x;
  const int tid = threadIdx.x;
  const float* P = xyz + (long long)b * N * 3;
  for (int i = tid; i < N; i += 256) dist[i] = 1e10f;
  if (tid == 0) sfar = 0;
  __syncthreads();
  for (int it = 0; it < npoint; ++it) {
    if (tid == 0) {
      int far = sfar;
      fidx[b * npoint + it] = far;
      cen[0] = P[far * 3 + 0]; cen[1] = P[far * 3 + 1]; cen[2] = P[far * 3 + 2];
    }
    __syncthreads();
    float bm = -1.0f; int bi = 0;
    for (int i = tid; i < N; i += 256) {
      float dx = P[i * 3 + 0] - cen[0];
      float dy = P[i * 3 + 1] - cen[1];
      float dz = P[i * 3 + 2] - cen[2];
      float d  = dx * dx + dy * dy + dz * dz;
      float dd = fminf(dist[i], d);
      dist[i] = dd;
      if (dd > bm) { bm = dd; bi = i; }
    }
    rmax[tid] = bm; rloc[tid] = bi;
    __syncthreads();
    for (int off = 128; off; off >>= 1) {
      if (tid < off) {
        if (rmax[tid + off] > rmax[tid] ||
            (rmax[tid + off] == rmax[tid] && rloc[tid + off] < rloc[tid])) {
          rmax[tid] = rmax[tid + off]; rloc[tid] = rloc[tid + off];
        }
      }
      __syncthreads();
    }
    if (tid == 0) sfar = rloc[0];
    __syncthreads();
  }
}

// ============================================================================
// Gather xyz rows by index: out[b,p,:] = xyz[b, idx[b,p], :]
// ============================================================================
__global__ void gather_xyz_kernel(const float* __restrict__ xyz, const int* __restrict__ idx,
                                  float* __restrict__ out, int B, int Np, int N) {
  long long t = (long long)blockIdx.x * 256 + threadIdx.x;
  if (t >= (long long)B * Np) return;
  int b = (int)(t / Np);
  long long s = ((long long)b * N + idx[t]) * 3;
  out[t * 3 + 0] = xyz[s + 0];
  out[t * 3 + 1] = xyz[s + 1];
  out[t * 3 + 2] = xyz[s + 2];
}

// ============================================================================
// kNN (k=32): one thread per query, insertion into sorted top-32 list.
// ============================================================================
__global__ void knn_kernel(const float* __restrict__ q, const float* __restrict__ r,
                           int* __restrict__ out, int B, int Nq, int Nr) {
  long long t = (long long)blockIdx.x * 256 + threadIdx.x;
  if (t >= (long long)B * Nq) return;
  int b = (int)(t / Nq);
  const float* Q = q + t * 3;
  const float* R = r + (long long)b * Nr * 3;
  float qx = Q[0], qy = Q[1], qz = Q[2];
  float bd[32]; int bi[32];
#pragma unroll
  for (int j = 0; j < 32; ++j) { bd[j] = 3.4e38f; bi[j] = 0; }
  for (int i = 0; i < Nr; ++i) {
    float dx = R[i * 3 + 0] - qx;
    float dy = R[i * 3 + 1] - qy;
    float dz = R[i * 3 + 2] - qz;
    float d = dx * dx + dy * dy + dz * dz;
    if (d < bd[31]) {
      int j = 31;
      while (j > 0 && bd[j - 1] > d) { bd[j] = bd[j - 1]; bi[j] = bi[j - 1]; --j; }
      bd[j] = d; bi[j] = i;
    }
  }
  for (int j = 0; j < 32; ++j) out[t * 32 + j] = bi[j];
}

// ============================================================================
// SA1 grouping: rel coords -> f16 rows of Kp=32 (3 real channels + zero pad)
// ============================================================================
__global__ void group_sa1_kernel(const float* __restrict__ xyz, const float* __restrict__ nxyz,
                                 const int* __restrict__ knn, _Float16* __restrict__ A, int B) {
  long long t = (long long)blockIdx.x * 256 + threadIdx.x;   // (b*512+p)*32 + j
  if (t >= (long long)B * 512 * 32) return;
  long long bp = t >> 5;
  int b = (int)(bp / 512);
  int ni = knn[t];
  const float* P = xyz + ((long long)b * 8192 + ni) * 3;
  const float* C = nxyz + bp * 3;
  _Float16* dst = A + t * 32;
  dst[0] = (_Float16)(P[0] - C[0]);
  dst[1] = (_Float16)(P[1] - C[1]);
  dst[2] = (_Float16)(P[2] - C[2]);
#pragma unroll
  for (int c = 3; c < 32; ++c) dst[c] = (_Float16)0.0f;
}

// ============================================================================
// SA2 grouping: [rel xyz (3) | gathered l1 features (128)] padded to Kp=160
// ============================================================================
__global__ void group_sa2_kernel(const float* __restrict__ l1_xyz, const float* __restrict__ l2_xyz,
                                 const int* __restrict__ knn, const float* __restrict__ l1_f,
                                 _Float16* __restrict__ A, int B) {
  long long t = (long long)blockIdx.x * 256 + threadIdx.x;   // (b*128+p)*32 + j
  if (t >= (long long)B * 128 * 32) return;
  long long bp = t >> 5;
  int b = (int)(bp / 128);
  int ni = knn[t];
  const float* P = l1_xyz + ((long long)b * 512 + ni) * 3;
  const float* C = l2_xyz + bp * 3;
  const float* F = l1_f + ((long long)b * 512 + ni) * 128;
  _Float16* dst = A + t * 160;
  dst[0] = (_Float16)(P[0] - C[0]);
  dst[1] = (_Float16)(P[1] - C[1]);
  dst[2] = (_Float16)(P[2] - C[2]);
  for (int c = 0; c < 128; ++c) dst[3 + c] = (_Float16)F[c];
#pragma unroll
  for (int c = 131; c < 160; ++c) dst[c] = (_Float16)0.0f;
}

// ============================================================================
// WMMA GEMM:  Y[M,N] = A[M,Kp](f16) x W[N,Kp](f16)^T (+bias), f32 accumulate.
// One wave computes one 16x16 tile via v_wmma_f32_16x16x32_f16.
// A-fragment lane l: row = l&15, K-halves [k0,k0+8) u [k0+16,k0+24), k0=(l>>4)*8.
// B-fragment = same striping over rows of W (since Y = A * W^T).
// D: lane l -> col l&15, VGPR r -> row r + 8*(l>>4).
// ============================================================================
__global__ __launch_bounds__(256)
void gemm_wmma_kernel(const _Float16* __restrict__ A, const _Float16* __restrict__ W,
                      float* __restrict__ Y, const float* __restrict__ bias,
                      int M, int Kp, int N) {
  const int lane = threadIdx.x & 31;
  const int wave = threadIdx.x >> 5;
  const int tilesN = N >> 4;
  const long long tile = (long long)blockIdx.x * 8 + wave;
  const long long tilesTot = (long long)(M >> 4) * tilesN;
  if (tile >= tilesTot) return;                 // wave-uniform: EXEC stays all-ones
  const int tm = (int)(tile / tilesN);
  const int tn = (int)(tile - (long long)tm * tilesN);
  const int k0 = (lane >> 4) * 8;
  const long long aBase = (long long)(tm * 16 + (lane & 15)) * Kp + k0;
  const long long bBase = (long long)(tn * 16 + (lane & 15)) * Kp + k0;
  v8f acc = {};
  for (int k = 0; k < Kp; k += 32) {
    v8h a0 = *(const v8h*)(A + aBase + k);
    v8h a1 = *(const v8h*)(A + aBase + k + 16);
    v8h b0 = *(const v8h*)(W + bBase + k);
    v8h b1 = *(const v8h*)(W + bBase + k + 16);
    v16h av, bv;
#pragma unroll
    for (int i = 0; i < 8; ++i) {
      av[i] = a0[i]; av[i + 8] = a1[i];
      bv[i] = b0[i]; bv[i + 8] = b1[i];
    }
    acc = __builtin_amdgcn_wmma_f32_16x16x32_f16(false, av, false, bv,
                                                 (short)0, acc, false, false);
  }
  const int col = tn * 16 + (lane & 15);
  const int rb  = tm * 16 + (lane >> 4) * 8;
  const float badd = bias ? bias[col] : 0.0f;
#pragma unroll
  for (int r = 0; r < 8; ++r)
    Y[(long long)(rb + r) * N + col] = acc[r] + badd;
}

// ============================================================================
// Per-channel mean / inv-std over M rows (deterministic tree reduce).
// stats[0..N) = mean, stats[N..2N) = rsqrt(var + eps)
// ============================================================================
__global__ __launch_bounds__(256)
void colstats_kernel(const float* __restrict__ Y, int M, int N, float* __restrict__ stats) {
  __shared__ float sh[256], sh2[256];
  const int n = blockIdx.x;
  const int tid = threadIdx.x;
  float s = 0.0f, s2 = 0.0f;
  for (int m = tid; m < M; m += 256) {
    float v = Y[(long long)m * N + n];
    s += v; s2 += v * v;
  }
  sh[tid] = s; sh2[tid] = s2;
  __syncthreads();
  for (int off = 128; off; off >>= 1) {
    if (tid < off) { sh[tid] += sh[tid + off]; sh2[tid] += sh2[tid + off]; }
    __syncthreads();
  }
  if (tid == 0) {
    float mean = sh[0] / (float)M;
    float var  = sh2[0] / (float)M - mean * mean;
    stats[n] = mean;
    stats[N + n] = rsqrtf(fmaxf(var, 0.0f) + EPS_BN);
  }
}

// ============================================================================
// BN + ReLU -> f16 (feeds next GEMM's A operand)
// ============================================================================
__global__ void bnrelu_kernel(const float* __restrict__ Y, const float* __restrict__ stats,
                              _Float16* __restrict__ dst, long long M, int N) {
  long long t = (long long)blockIdx.x * 256 + threadIdx.x;
  if (t >= M * N) return;
  int c = (int)(t % N);
  float v = fmaxf((Y[t] - stats[c]) * stats[N + c], 0.0f);
  dst[t] = (_Float16)v;
}

// ============================================================================
// Fused BN + ReLU + max-pool over k=32 neighbors (SA stage output)
// out[bp, c] = max_j relu((Y[(bp*32+j), c] - mean) * inv)
// ============================================================================
__global__ void bn_relu_pool_kernel(const float* __restrict__ Y, const float* __restrict__ stats,
                                    float* __restrict__ out, long long BP, int C) {
  long long t = (long long)blockIdx.x * 256 + threadIdx.x;
  if (t >= BP * C) return;
  long long bp = t / C;
  int c = (int)(t - bp * C);
  float mean = stats[c], inv = stats[C + c];
  float m = 0.0f;                              // relu output >= 0
  for (int j = 0; j < 32; ++j) {
    float v = fmaxf((Y[((bp << 5) + j) * (long long)C + c] - mean) * inv, 0.0f);
    m = fmaxf(m, v);
  }
  out[t] = m;
}

// ============================================================================
// 3-NN interpolation weights: per dst point, top-3 nearest src + idw weights
// ============================================================================
__global__ void interp_weights_kernel(const float* __restrict__ dst_xyz,
                                      const float* __restrict__ src_xyz,
                                      int B, int Nd, int Ns,
                                      int* __restrict__ idx3, float* __restrict__ w3) {
  long long t = (long long)blockIdx.x * 256 + threadIdx.x;
  if (t >= (long long)B * Nd) return;
  int b = (int)(t / Nd);
  const float* D = dst_xyz + t * 3;
  const float* S = src_xyz + (long long)b * Ns * 3;
  float qx = D[0], qy = D[1], qz = D[2];
  float bd[3] = {3.4e38f, 3.4e38f, 3.4e38f};
  int   bi[3] = {0, 0, 0};
  for (int i = 0; i < Ns; ++i) {
    float dx = S[i * 3 + 0] - qx;
    float dy = S[i * 3 + 1] - qy;
    float dz = S[i * 3 + 2] - qz;
    float d = dx * dx + dy * dy + dz * dz;
    if (d < bd[2]) {
      int j = 2;
      while (j > 0 && bd[j - 1] > d) { bd[j] = bd[j - 1]; bi[j] = bi[j - 1]; --j; }
      bd[j] = d; bi[j] = i;
    }
  }
  float w[3], ws = 0.0f;
#pragma unroll
  for (int j = 0; j < 3; ++j) {
    float d = sqrtf(fmaxf(bd[j], 0.0f));
    w[j] = 1.0f / (d + 1e-8f);
    ws += w[j];
  }
#pragma unroll
  for (int j = 0; j < 3; ++j) { idx3[t * 3 + j] = bi[j]; w3[t * 3 + j] = w[j] / ws; }
}

// ============================================================================
// FP2 concat: A[row, 0:128] = l1_f, A[row, 128:384] = interp(l2_f); f16 out
// ============================================================================
__global__ void fp2_concat_kernel(const float* __restrict__ l1_f, const float* __restrict__ l2_f,
                                  const int* __restrict__ idx3, const float* __restrict__ w3,
                                  _Float16* __restrict__ A, int B) {
  long long t = (long long)blockIdx.x * 256 + threadIdx.x;   // over B*512*384
  if (t >= (long long)B * 512 * 384) return;
  long long row = t / 384;
  int c = (int)(t - row * 384);
  int b = (int)(row / 512);
  float v;
  if (c < 128) {
    v = l1_f[row * 128 + c];
  } else {
    int cc = c - 128;
    v = 0.0f;
#pragma unroll
    for (int j = 0; j < 3; ++j)
      v += l2_f[((long long)b * 128 + idx3[row * 3 + j]) * 256 + cc] * w3[row * 3 + j];
  }
  A[t] = (_Float16)v;
}

// ============================================================================
// FP1 interp features (src features are f16 l1_up): f16 out rows of 128
// ============================================================================
__global__ void fp1_interp_kernel(const _Float16* __restrict__ srcF,
                                  const int* __restrict__ idx3, const float* __restrict__ w3,
                                  _Float16* __restrict__ A, int B) {
  long long t = (long long)blockIdx.x * 256 + threadIdx.x;   // over B*8192*128
  if (t >= (long long)B * 8192 * 128) return;
  long long row = t >> 7;
  int c = (int)(t & 127);
  int b = (int)(row / 8192);
  float v = 0.0f;
#pragma unroll
  for (int j = 0; j < 3; ++j)
    v += (float)srcF[((long long)b * 512 + idx3[row * 3 + j]) * 128 + c] * w3[row * 3 + j];
  A[t] = (_Float16)v;
}

// ============================================================================
// Final L2 row-normalize (64 channels) -> d_out
// ============================================================================
__global__ void l2norm_kernel(const float* __restrict__ Y, float* __restrict__ out, long long M) {
  long long r = (long long)blockIdx.x * 256 + threadIdx.x;
  if (r >= M) return;
  float s = 0.0f;
#pragma unroll
  for (int c = 0; c < 64; ++c) { float v = Y[r * 64 + c]; s += v * v; }
  float inv = 1.0f / fmaxf(sqrtf(s), 1e-12f);
#pragma unroll
  for (int c = 0; c < 64; ++c) out[r * 64 + c] = Y[r * 64 + c] * inv;
}

// ============================================================================
// Host orchestration
// ============================================================================
static inline int cdiv(long long a, long long b) { return (int)((a + b - 1) / b); }

extern "C" void kernel_launch(void* const* d_in, const int* in_sizes, int n_in,
                              void* d_out, int out_size, void* d_ws, size_t ws_size,
                              hipStream_t stream) {
  (void)in_sizes; (void)n_in; (void)out_size; (void)ws_size;
  const int B = 16, N0 = 8192;

  const float* x      = (const float*)d_in[0];
  const float* w_s1[3] = {(const float*)d_in[1], (const float*)d_in[2], (const float*)d_in[3]};
  const float* w_s2[3] = {(const float*)d_in[4], (const float*)d_in[5], (const float*)d_in[6]};
  const float* w_f2[2] = {(const float*)d_in[7], (const float*)d_in[8]};
  const float* w_f1[3] = {(const float*)d_in[9], (const float*)d_in[10], (const float*)d_in[11]};
  const float* hw0 = (const float*)d_in[12];
  const float* hw1 = (const float*)d_in[13];
  const float* hb1 = (const float*)d_in[14];

  // --- bump allocator over workspace ---
  char* base = (char*)d_ws;
  size_t off = 0;
  auto alloc = [&](size_t bytes) -> void* {
    off = (off + 255) & ~(size_t)255;
    void* p = base + off;
    off += bytes;
    return p;
  };

  // padded f16 weights: {N, K, Kp}
  struct WDesc { const float* src; int N, K, Kp; _Float16* dst; };
  WDesc wd[13] = {
    {w_s1[0],  64,   3,  32, nullptr}, {w_s1[1],  64,  64,  64, nullptr},
    {w_s1[2], 128,  64,  64, nullptr}, {w_s2[0], 128, 131, 160, nullptr},
    {w_s2[1], 128, 128, 128, nullptr}, {w_s2[2], 256, 128, 128, nullptr},
    {w_f2[0], 256, 384, 384, nullptr}, {w_f2[1], 128, 256, 256, nullptr},
    {w_f1[0], 128, 128, 128, nullptr}, {w_f1[1], 128, 128, 128, nullptr},
    {w_f1[2], 128, 128, 128, nullptr}, {hw0,     128, 128, 128, nullptr},
    {hw1,      64, 128, 128, nullptr},
  };
  for (int i = 0; i < 13; ++i) {
    wd[i].dst = (_Float16*)alloc((size_t)wd[i].N * wd[i].Kp * 2);
    int tot = wd[i].N * wd[i].Kp;
    padw_kernel<<<cdiv(tot, 256), 256, 0, stream>>>(wd[i].src, wd[i].dst, wd[i].N, wd[i].K, wd[i].Kp);
  }

  int*   fidx1  = (int*)  alloc((size_t)B * 512 * 4);
  float* l1_xyz = (float*)alloc((size_t)B * 512 * 3 * 4);
  int*   knn1   = (int*)  alloc((size_t)B * 512 * 32 * 4);
  int*   fidx2  = (int*)  alloc((size_t)B * 128 * 4);
  float* l2_xyz = (float*)alloc((size_t)B * 128 * 3 * 4);
  int*   knn2   = (int*)  alloc((size_t)B * 128 * 32 * 4);
  float* stats  = (float*)alloc(2 * 256 * 4);
  float* l1_f   = (float*)alloc((size_t)B * 512 * 128 * 4);
  float* l2_f   = (float*)alloc((size_t)B * 128 * 256 * 4);
  int*   iw2i   = (int*)  alloc((size_t)B * 512 * 3 * 4);
  float* iw2w   = (float*)alloc((size_t)B * 512 * 3 * 4);
  int*   iw1i   = (int*)  alloc((size_t)B * 8192 * 3 * 4);
  float* iw1w   = (float*)alloc((size_t)B * 8192 * 3 * 4);
  _Float16* A0  = (_Float16*)alloc((size_t)33554432);   // 16.7M halves
  _Float16* A1  = (_Float16*)alloc((size_t)33554432);
  float* Y      = (float*)alloc((size_t)262144 * 128 * 4);  // 134 MB

  auto gemm = [&](const _Float16* A, const _Float16* W, float* y, int M, int Kp, int Nn,
                  const float* bias) {
    long long tiles = (long long)(M / 16) * (Nn / 16);
    gemm_wmma_kernel<<<cdiv(tiles, 8), 256, 0, stream>>>(A, W, y, bias, M, Kp, Nn);
  };
  auto bnrelu = [&](float* y, int M, int Nn, _Float16* dst) {
    colstats_kernel<<<Nn, 256, 0, stream>>>(y, M, Nn, stats);
    bnrelu_kernel<<<cdiv((long long)M * Nn, 256), 256, 0, stream>>>(y, stats, dst, M, Nn);
  };
  auto bnpool = [&](float* y, long long BP, int C, float* out) {
    colstats_kernel<<<C, 256, 0, stream>>>(y, (int)(BP * 32), C, stats);
    bn_relu_pool_kernel<<<cdiv(BP * C, 256), 256, 0, stream>>>(y, stats, out, BP, C);
  };

  // ---------------- SA1 ----------------
  fps_kernel<<<B, 256, 0, stream>>>(x, fidx1, N0, 512);
  gather_xyz_kernel<<<cdiv((long long)B * 512, 256), 256, 0, stream>>>(x, fidx1, l1_xyz, B, 512, N0);
  knn_kernel<<<cdiv((long long)B * 512, 256), 256, 0, stream>>>(l1_xyz, x, knn1, B, 512, N0);
  group_sa1_kernel<<<cdiv((long long)B * 512 * 32, 256), 256, 0, stream>>>(x, l1_xyz, knn1, A0, B);
  gemm(A0, wd[0].dst, Y, 262144, 32, 64, nullptr);  bnrelu(Y, 262144, 64, A1);
  gemm(A1, wd[1].dst, Y, 262144, 64, 64, nullptr);  bnrelu(Y, 262144, 64, A0);
  gemm(A0, wd[2].dst, Y, 262144, 64, 128, nullptr); bnpool(Y, (long long)B * 512, 128, l1_f);

  // ---------------- SA2 ----------------
  fps_kernel<<<B, 256, 0, stream>>>(l1_xyz, fidx2, 512, 128);
  gather_xyz_kernel<<<cdiv((long long)B * 128, 256), 256, 0, stream>>>(l1_xyz, fidx2, l2_xyz, B, 128, 512);
  knn_kernel<<<cdiv((long long)B * 128, 256), 256, 0, stream>>>(l2_xyz, l1_xyz, knn2, B, 128, 512);
  group_sa2_kernel<<<cdiv((long long)B * 128 * 32, 256), 256, 0, stream>>>(l1_xyz, l2_xyz, knn2, l1_f, A0, B);
  gemm(A0, wd[3].dst, Y, 65536, 160, 128, nullptr); bnrelu(Y, 65536, 128, A1);
  gemm(A1, wd[4].dst, Y, 65536, 128, 128, nullptr); bnrelu(Y, 65536, 128, A0);
  gemm(A0, wd[5].dst, Y, 65536, 128, 256, nullptr); bnpool(Y, (long long)B * 128, 256, l2_f);

  // ---------------- FP2 ----------------
  interp_weights_kernel<<<cdiv((long long)B * 512, 256), 256, 0, stream>>>(l1_xyz, l2_xyz, B, 512, 128, iw2i, iw2w);
  fp2_concat_kernel<<<cdiv((long long)B * 512 * 384, 256), 256, 0, stream>>>(l1_f, l2_f, iw2i, iw2w, A0, B);
  gemm(A0, wd[6].dst, Y, 8192, 384, 256, nullptr); bnrelu(Y, 8192, 256, A1);
  gemm(A1, wd[7].dst, Y, 8192, 256, 128, nullptr); bnrelu(Y, 8192, 128, A0);   // A0 = l1_up (f16)

  // ---------------- FP1 ----------------
  interp_weights_kernel<<<cdiv((long long)B * 8192, 256), 256, 0, stream>>>(x, l1_xyz, B, 8192, 512, iw1i, iw1w);
  fp1_interp_kernel<<<cdiv((long long)B * 8192 * 128, 256), 256, 0, stream>>>(A0, iw1i, iw1w, A1, B);
  gemm(A1, wd[8].dst,  Y, 131072, 128, 128, nullptr); bnrelu(Y, 131072, 128, A0);
  gemm(A0, wd[9].dst,  Y, 131072, 128, 128, nullptr); bnrelu(Y, 131072, 128, A1);
  gemm(A1, wd[10].dst, Y, 131072, 128, 128, nullptr); bnrelu(Y, 131072, 128, A0);

  // ---------------- Head ----------------
  gemm(A0, wd[11].dst, Y, 131072, 128, 128, nullptr); bnrelu(Y, 131072, 128, A1);
  gemm(A1, wd[12].dst, Y, 131072, 128, 64, hb1);
  l2norm_kernel<<<cdiv((long long)131072, 256), 256, 0, stream>>>(Y, (float*)d_out, 131072);
}
